// LSTMCRFModel_27599459844409
// MI455X (gfx1250) — compile-verified, matrix operand order checked by reading, not run on hardware
//
#include <hip/hip_runtime.h>
#include <stdint.h>

#define V_ 50000
#define E_ 256
#define H_ 256
#define L_ 9
#define B_ 64
#define T_ 512

typedef __attribute__((ext_vector_type(16))) __bf16   v16bf;
typedef __attribute__((ext_vector_type(8)))  float    v8f;
typedef __attribute__((ext_vector_type(8)))  uint32_t v8u;
typedef __attribute__((ext_vector_type(4)))  uint32_t v4u;
typedef __attribute__((ext_vector_type(8)))  int      v8i;
typedef __attribute__((ext_vector_type(4)))  int      v4i;

// ---- helpers ---------------------------------------------------------------

// Load one 16x32 bf16 WMMA fragment (A or B) from a row of packed bf16 pairs.
// ISA 7.12.2 16-bit 16x32 layout: lanes 0-15 -> K pairs {0..7 | 16..23},
// lanes 16-31 -> K pairs {8..15 | 24..31}; half = lane>>4 selects the shift.
static __device__ __forceinline__ v16bf frag_from_dw(const uint32_t* dw, int half) {
  v8u u;
  const int o = half * 4;
  u[0] = dw[o + 0]; u[1] = dw[o + 1]; u[2] = dw[o + 2]; u[3] = dw[o + 3];
  u[4] = dw[o + 8]; u[5] = dw[o + 9]; u[6] = dw[o + 10]; u[7] = dw[o + 11];
  return __builtin_bit_cast(v16bf, u);
}

static __device__ __forceinline__ uint32_t pack_bf2(float a, float b) {
  union { __bf16 h; uint16_t u; } x, y;
  x.h = (__bf16)a; y.h = (__bf16)b;
  return ((uint32_t)y.u << 16) | (uint32_t)x.u;
}

static __device__ __forceinline__ float sigm(float x) {
  return 1.f / (1.f + __expf(-x));
}

// Tensor Data Mover: async-load a 16x256 bf16 tile (rows contiguous, 8 KB)
// from global memory into LDS at lds_addr. D# per CDNA5 ISA ch.8.
// 6-arg builtin form (clang-23 / therock-10.0 toolchain).
static __device__ __forceinline__ void tdm_load_x(uint32_t lds_addr, const uint32_t* gptr) {
  const uint64_t ga = (uint64_t)(uintptr_t)gptr;
  v4u g0;
  g0[0] = 1u;                                               // count=1, user D#
  g0[1] = lds_addr;                                         // LDS byte address
  g0[2] = (uint32_t)ga;                                     // global_addr[31:0]
  g0[3] = (uint32_t)((ga >> 32) & 0x1FFFFFFu) | (2u << 30); // addr[56:32] | type=2
  v8i g1;
  g1[0] = (int)(1u << 16);        // data_size=1 -> 2 bytes/elem
  g1[1] = (int)(256u << 16);      // tensor_dim0 = 256 (low 16 in bits 63:48)
  g1[2] = (int)(16u << 16);       // tensor_dim1 = 16  (low 16 in bits 95:80)
  g1[3] = (int)(256u << 16);      // tile_dim0 = 256 (bits 127:112)
  g1[4] = (int)16;                // tile_dim1 = 16, tile_dim2 = 0
  g1[5] = (int)256;               // tensor_dim0_stride = 256 elems
  g1[6] = 0;
  g1[7] = 0;
  const v4i z4 = {0, 0, 0, 0};
  const v8i z8 = {0, 0, 0, 0, 0, 0, 0, 0};
  __builtin_amdgcn_tensor_load_to_lds(g0, g1, z4, z4, z8, 0);
}

// ---- kernel 1: cast/concat LSTM weights to bf16 [dir][1024][512] -----------

__global__ void prep_w_kernel(const float* __restrict__ wihf, const float* __restrict__ whhf,
                              const float* __restrict__ wihb, const float* __restrict__ whhb,
                              uint32_t* __restrict__ wcat) {
  int i = blockIdx.x * blockDim.x + threadIdx.x;      // dword index, 2*1024*256 total
  if (i >= 2 * 1024 * 256) return;
  int d   = i >> 18;
  int rem = i & ((1 << 18) - 1);
  int g   = rem >> 8;
  int k   = (rem & 255) * 2;                          // element index in [0,512)
  const float* wih = d ? wihb : wihf;
  const float* whh = d ? whhb : whhf;
  float v0, v1;
  if (k < 256) { v0 = wih[g * 256 + k];       v1 = wih[g * 256 + k + 1]; }
  else         { v0 = whh[g * 256 + k - 256]; v1 = whh[g * 256 + k - 255]; }
  wcat[i] = pack_bf2(v0, v1);
}

// ---- kernel 2: biases, padded FC weights, zero output ----------------------

__global__ void prep_small_kernel(const float* __restrict__ bihf, const float* __restrict__ bhhf,
                                  const float* __restrict__ bihb, const float* __restrict__ bhhb,
                                  const float* __restrict__ fcw,  const float* __restrict__ fcb,
                                  float* __restrict__ bias, uint32_t* __restrict__ fcwp,
                                  float* __restrict__ fcbp, float* __restrict__ out) {
  const int tid = threadIdx.x;                         // 1024 threads, 1 block
  if (tid == 0) out[0] = 0.f;
  for (int i = tid; i < 2048; i += 1024) {
    int d = i >> 10, g = i & 1023;
    bias[i] = d ? (bihb[g] + bhhb[g]) : (bihf[g] + bhhf[g]);
  }
  for (int i = tid; i < 16 * 256; i += 1024) {         // dwords of [16][512] bf16
    int l = i >> 8, k = (i & 255) * 2;
    float v0 = (l < L_) ? fcw[l * 512 + k]     : 0.f;
    float v1 = (l < L_) ? fcw[l * 512 + k + 1] : 0.f;
    fcwp[i] = pack_bf2(v0, v1);
  }
  if (tid < 16) fcbp[tid] = (tid < L_) ? fcb[tid] : 0.f;
}

// ---- kernel 3: embedding gather + bf16 cast, time-major [T][B][E] ----------

__global__ void embed_kernel(const int* __restrict__ ids, const float* __restrict__ emb,
                             uint32_t* __restrict__ xbf) {
  size_t n = (size_t)blockIdx.x * blockDim.x + threadIdx.x;  // 8 elems each
  size_t base = n * 8;
  if (base >= (size_t)T_ * B_ * E_) return;
  int e = (int)(base & 255);
  int b = (int)((base >> 8) & 63);
  int t = (int)(base >> 14);
  int id = ids[b * T_ + t];
  const float* src = emb + (size_t)id * E_ + e;
  uint32_t* dst = xbf + (base >> 1);
#pragma unroll
  for (int q = 0; q < 4; ++q) dst[q] = pack_bf2(src[2 * q], src[2 * q + 1]);
}

// ---- kernel 4: persistent bf16-WMMA LSTM recurrence ------------------------
// grid (4 batch-chunks, 2 directions), 1024 threads = 32 waves.
// Per step: gates(16x1024) = [x_t | h](16x512) @ Wcat^T(512x1024) via WMMA.
// x_t tiles are TDM-prefetched (double-buffered) into LDS one step ahead;
// W streams from L2 every step via global_load (offset laundered per step so
// LICM cannot hoist the loads and spill them, but addrspace stays global).

__global__ __launch_bounds__(1024) void lstm_kernel(
    const uint32_t* __restrict__ xbf, const uint32_t* __restrict__ wcat,
    const float* __restrict__ bias, uint32_t* __restrict__ hcat) {
  __shared__ float    gates[16][1024];                 // 64 KB
  __shared__ uint32_t hbf[16][128];                    //  8 KB (16 x 256 bf16)
  __shared__ uint32_t xs[2][16][128];                  // 16 KB x-tile double buffer

  const int bc   = blockIdx.x;                         // batch chunk (16 rows)
  const int dir  = blockIdx.y;                         // 0=fwd, 1=bwd
  const int tid  = threadIdx.x;
  const int lane = tid & 31, wave = tid >> 5;
  const int half = lane >> 4, idx = lane & 15;

  for (int i = tid; i < 16 * 128; i += 1024) ((uint32_t*)hbf)[i] = 0u;

  // elementwise mapping: each thread owns 4 consecutive hidden units
  const int erow = tid >> 6;
  const int ej   = (tid & 63) * 4;
  float creg[4] = {0.f, 0.f, 0.f, 0.f};
  const float* bd = bias + dir * 1024;
  float bI[4], bF[4], bG[4], bO[4];
#pragma unroll
  for (int q = 0; q < 4; ++q) {
    bI[q] = bd[ej + q];       bF[q] = bd[256 + ej + q];
    bG[q] = bd[512 + ej + q]; bO[q] = bd[768 + ej + q];
  }

  const uint32_t* wd = wcat + (size_t)dir * 1024 * 256;   // rows of 256 dwords
  const int n0a = (wave * 2 + 0) * 16;
  const int n0b = (wave * 2 + 1) * 16;
  const uint32_t* wrowA = wd + (size_t)(n0a + idx) * 256;
  const uint32_t* wrowB = wd + (size_t)(n0b + idx) * 256;

  const uint32_t lds_xs[2] = { (uint32_t)(uintptr_t)&xs[0][0][0],
                               (uint32_t)(uintptr_t)&xs[1][0][0] };

  // TDM-prefetch the first x tile (wave 0 only; TDM is a per-wave DMA op)
  if (wave == 0) {
    const int t0 = dir ? (T_ - 1) : 0;
    tdm_load_x(lds_xs[0], xbf + ((size_t)t0 * B_ + bc * 16) * 128);
  }

  for (int s = 0; s < T_; ++s) {
    const int t   = dir ? (T_ - 1 - s) : s;
    const int buf = s & 1;
    __builtin_amdgcn_s_wait_tensorcnt(0);              // x tile in LDS (wave 0)
    __syncthreads();                                   // x + prev h visible to all
    if (wave == 0 && s + 1 < T_) {                     // prefetch next x tile
      const int tn = dir ? (T_ - 2 - s) : (s + 1);
      tdm_load_x(lds_xs[buf ^ 1], xbf + ((size_t)tn * B_ + bc * 16) * 128);
    }

    // launder a zero offset each step: keeps W loads inside the loop (no
    // LICM hoist/spill) while the pointer chain stays global-addrspace.
    uint32_t woff = 0;
    asm volatile("" : "+v"(woff));
    const uint32_t* wA = wrowA + woff;
    const uint32_t* wB = wrowB + woff;

    v8f acc0 = {}; v8f acc1 = {};
#pragma unroll
    for (int kc = 0; kc < 16; ++kc) {
      v16bf a;
      if (kc < 8) {                                    // K 0..255 = x_t (LDS)
        v8u u; const int o = kc * 16 + half * 4;
        u[0] = xs[buf][idx][o + 0];  u[1] = xs[buf][idx][o + 1];
        u[2] = xs[buf][idx][o + 2];  u[3] = xs[buf][idx][o + 3];
        u[4] = xs[buf][idx][o + 8];  u[5] = xs[buf][idx][o + 9];
        u[6] = xs[buf][idx][o + 10]; u[7] = xs[buf][idx][o + 11];
        a = __builtin_bit_cast(v16bf, u);
      } else {                                         // K 256..511 = h (LDS)
        v8u u; const int o = (kc - 8) * 16 + half * 4;
        u[0] = hbf[idx][o + 0];  u[1] = hbf[idx][o + 1];
        u[2] = hbf[idx][o + 2];  u[3] = hbf[idx][o + 3];
        u[4] = hbf[idx][o + 8];  u[5] = hbf[idx][o + 9];
        u[6] = hbf[idx][o + 10]; u[7] = hbf[idx][o + 11];
        a = __builtin_bit_cast(v16bf, u);
      }
      v16bf b0f = frag_from_dw(wA + kc * 16, half);
      v16bf b1f = frag_from_dw(wB + kc * 16, half);
      acc0 = __builtin_amdgcn_wmma_f32_16x16x32_bf16(false, a, false, b0f, (short)0, acc0, false, false);
      acc1 = __builtin_amdgcn_wmma_f32_16x16x32_bf16(false, a, false, b1f, (short)0, acc1, false, false);
    }
#pragma unroll
    for (int r = 0; r < 8; ++r) {
      gates[r + half * 8][n0a + idx] = acc0[r];
      gates[r + half * 8][n0b + idx] = acc1[r];
    }
    __syncthreads();                                   // gates complete

    float hv[4];
#pragma unroll
    for (int q = 0; q < 4; ++q) {
      float ig = gates[erow][      ej + q] + bI[q];
      float fg = gates[erow][256 + ej + q] + bF[q];
      float gg = gates[erow][512 + ej + q] + bG[q];
      float og = gates[erow][768 + ej + q] + bO[q];
      float cn = sigm(fg) * creg[q] + sigm(ig) * tanhf(gg);
      creg[q]  = cn;
      hv[q]    = sigm(og) * tanhf(cn);
    }
    const uint32_t hp0 = pack_bf2(hv[0], hv[1]);
    const uint32_t hp1 = pack_bf2(hv[2], hv[3]);
    hbf[erow][(ej >> 1) + 0] = hp0;
    hbf[erow][(ej >> 1) + 1] = hp1;
    uint32_t* ho = hcat + ((size_t)t * B_ + bc * 16 + erow) * 256 + dir * 128 + (ej >> 1);
    ho[0] = hp0; ho[1] = hp1;
    // loop-top barrier orders these h writes before the next step's reads
  }
}

// ---- kernel 5: FC projection via WMMA, logits [T*B][16] --------------------

__global__ __launch_bounds__(256) void fc_kernel(const uint32_t* __restrict__ hcat,
                                                 const uint32_t* __restrict__ fcwp,
                                                 const float* __restrict__ fcbp,
                                                 float* __restrict__ logits) {
  const int tid = threadIdx.x, lane = tid & 31, wave = tid >> 5;
  const int half = lane >> 4, idx = lane & 15;
  const int m0 = (blockIdx.x * 8 + wave) * 16;
  const uint32_t* arow = hcat + (size_t)(m0 + idx) * 256;
  const uint32_t* brow = fcwp + (size_t)idx * 256;
  v8f acc = {};
#pragma unroll
  for (int kc = 0; kc < 16; ++kc) {
    v16bf a = frag_from_dw(arow + kc * 16, half);
    v16bf b = frag_from_dw(brow + kc * 16, half);
    acc = __builtin_amdgcn_wmma_f32_16x16x32_bf16(false, a, false, b, (short)0, acc, false, false);
  }
  const float bb = fcbp[idx];
#pragma unroll
  for (int r = 0; r < 8; ++r)
    logits[(size_t)(m0 + r + half * 8) * 16 + idx] = acc[r] + bb;
}

// ---- kernel 6: CRF numerator + forward logZ, 16 lanes per sequence ---------

__global__ __launch_bounds__(256) void crf_kernel(const int* __restrict__ ids,
                                                  const int* __restrict__ labels,
                                                  const float* __restrict__ logits,
                                                  const float* __restrict__ startv,
                                                  const float* __restrict__ endv,
                                                  const float* __restrict__ transg,
                                                  float* __restrict__ out) {
  __shared__ float tr[81];
  const int tid = threadIdx.x;
  if (tid < 81) tr[tid] = transg[tid];
  __syncthreads();
  const int b     = blockIdx.x * 16 + (tid >> 4);
  const int j     = tid & 15;
  const int lane  = tid & 31;
  const int gbase = lane & 16;                          // 16-lane group base in wave
  const bool act  = (j < L_);
  const int  jj   = act ? j : 0;

  float alpha = act ? (startv[j] + logits[(size_t)b * 16 + j]) : -1e30f;
  float num = 0.f; int count = 0; int prevTag = 0;
  if (j == 0) {
    prevTag = labels[b * T_];
    num = startv[prevTag] + logits[(size_t)b * 16 + prevTag];
    count = (ids[b * T_] != 0) ? 1 : 0;
  }

  for (int t = 1; t < T_; ++t) {
    const int maskt = (ids[b * T_ + t] != 0);
    const float em = logits[((size_t)t * B_ + b) * 16 + jj];
    float av[L_]; float m = -1e30f;
#pragma unroll
    for (int i = 0; i < L_; ++i) {
      float ai = __shfl(alpha, gbase + i, 32);
      av[i] = ai + tr[i * L_ + jj];
      m = fmaxf(m, av[i]);
    }
    float ssum = 0.f;
#pragma unroll
    for (int i = 0; i < L_; ++i) ssum += __expf(av[i] - m);
    const float nxt = m + __logf(ssum) + em;
    if (act && maskt) alpha = nxt;
    if (j == 0) {
      const int tg = labels[b * T_ + t];
      if (maskt) {
        num += tr[prevTag * L_ + tg] + logits[((size_t)t * B_ + b) * 16 + tg];
        ++count;
      }
      prevTag = tg;
    }
  }
  if (j == 0) {
    const int lastTag = labels[b * T_ + (count > 0 ? count - 1 : 0)];
    num += endv[lastTag];
  }
  float v = act ? (alpha + endv[j]) : -1e30f;
  float m = v;
#pragma unroll
  for (int off = 8; off >= 1; off >>= 1) m = fmaxf(m, __shfl_xor(m, off, 32));
  float s = __expf(v - m);
#pragma unroll
  for (int off = 8; off >= 1; off >>= 1) s += __shfl_xor(s, off, 32);
  const float logZ = m + __logf(s);
  if (j == 0) atomicAdd(out, logZ - num);               // loss = sum(logZ - num)
}

// ---- host launcher ---------------------------------------------------------

extern "C" void kernel_launch(void* const* d_in, const int* in_sizes, int n_in,
                              void* d_out, int out_size, void* d_ws, size_t ws_size,
                              hipStream_t stream) {
  (void)in_sizes; (void)n_in; (void)out_size; (void)ws_size;
  const int*   ids    = (const int*)  d_in[0];
  const int*   labels = (const int*)  d_in[1];
  const float* emb    = (const float*)d_in[2];
  const float* wihf   = (const float*)d_in[3];
  const float* whhf   = (const float*)d_in[4];
  const float* bihf   = (const float*)d_in[5];
  const float* bhhf   = (const float*)d_in[6];
  const float* wihb   = (const float*)d_in[7];
  const float* whhb   = (const float*)d_in[8];
  const float* bihb   = (const float*)d_in[9];
  const float* bhhb   = (const float*)d_in[10];
  const float* fcw    = (const float*)d_in[11];
  const float* fcb    = (const float*)d_in[12];
  const float* crfS   = (const float*)d_in[13];
  const float* crfE   = (const float*)d_in[14];
  const float* crfT   = (const float*)d_in[15];
  float* out = (float*)d_out;

  uint8_t* p = (uint8_t*)d_ws;
  auto take = [&](size_t bytes) {
    uint8_t* q = p;
    p += (bytes + 255) & ~(size_t)255;
    return q;
  };
  uint32_t* wcat   = (uint32_t*)take((size_t)2 * 1024 * 512 * 2);   //  2.0 MB bf16
  float*    bias   = (float*)   take((size_t)2 * 1024 * 4);
  uint32_t* fcwp   = (uint32_t*)take((size_t)16 * 512 * 2);
  float*    fcbp   = (float*)   take((size_t)16 * 4);
  uint32_t* xbf    = (uint32_t*)take((size_t)T_ * B_ * E_ * 2);     // 16.8 MB bf16
  uint32_t* hcat   = (uint32_t*)take((size_t)T_ * B_ * 512 * 2);    // 33.6 MB bf16
  float*    logits = (float*)   take((size_t)T_ * B_ * 16 * 4);     //  2.1 MB f32

  prep_w_kernel    <<<2048, 256, 0, stream>>>(wihf, whhf, wihb, whhb, wcat);
  prep_small_kernel<<<1, 1024, 0, stream>>>(bihf, bhhf, bihb, bhhb, fcw, fcb,
                                            bias, fcwp, fcbp, out);
  embed_kernel     <<<4096, 256, 0, stream>>>(ids, emb, xbf);
  lstm_kernel      <<<dim3(4, 2), 1024, 0, stream>>>(xbf, wcat, bias, hcat);
  fc_kernel        <<<256, 256, 0, stream>>>(hcat, fcwp, fcbp, logits);
  crf_kernel       <<<4, 256, 0, stream>>>(ids, labels, logits, crfS, crfE, crfT, out);
}
